// MultiHeadRelationalModule_798863917498
// MI455X (gfx1250) — matrix operand order repaired and checked by reading, not compiled, not run on hardware
//
#include <hip/hip_runtime.h>
#include <hip/hip_bf16.h>
#include <math.h>

#define B_     16
#define N_     1024
#define HEADS_ 4
#define D_     64
#define FEAT_  26
#define BH_    64            // B_*HEADS_
#define CNT1_  262144        // HEADS_*N_*D_
#define CNT2_  65536         // N_*D_
#define FRAG_  512           // ushorts per 16x32 (A) or 32x16 (B) bf16 fragment

typedef __attribute__((ext_vector_type(16))) __bf16         v16bf;
typedef __attribute__((ext_vector_type(8)))  float          v8f;
typedef __attribute__((ext_vector_type(16))) unsigned short v16us;

__device__ __forceinline__ unsigned short f2bf(float f) {
    unsigned u = __float_as_uint(f);
    unsigned r = u + 0x7fffu + ((u >> 16) & 1u);   // round-to-nearest-even
    return (unsigned short)(r >> 16);
}
__device__ __forceinline__ float bf2f(unsigned short s) {
    return __uint_as_float(((unsigned)s) << 16);
}

__device__ __forceinline__ v8f wmma_acc(v16us a, v16us b, v8f acc) {
    return __builtin_amdgcn_wmma_f32_16x16x32_bf16(
        false, __builtin_bit_cast(v16bf, a),
        false, __builtin_bit_cast(v16bf, b),
        (short)0, acc, false, false);
}

// ---------------- stage 0: xin = concat(x, coords), [B,N,26] ----------------
__global__ void k_xin(const float* __restrict__ x, float* __restrict__ xin) {
    int g = blockIdx.x * 256 + threadIdx.x;          // 16384
    int b = g >> 10, n = g & 1023, hy = n >> 5, wx = n & 31;
    float* op = xin + (size_t)g * FEAT_;
    const float* xp = x + (((size_t)b * 24) * 32 + hy) * 32 + wx;
    #pragma unroll
    for (int f = 0; f < 24; f++) op[f] = xp[(size_t)f * 1024];
    op[24] = (float)wx * (1.0f / 32.0f);
    op[25] = (float)hy * (1.0f / 32.0f);
}

// ---------------- projections: out[b,h,n,d] = xin@w^T + bias ----------------
__global__ void k_proj(const float* __restrict__ xin, const float* __restrict__ w,
                       const float* __restrict__ bias, float* __restrict__ out) {
    int g = blockIdx.x * 256 + threadIdx.x;          // 16*1024*256
    int o = g & 255, bn = g >> 8;
    const float* xp = xin + (size_t)bn * FEAT_;
    const float* wp = w + (size_t)o * FEAT_;
    float s = bias[o];
    #pragma unroll
    for (int f = 0; f < FEAT_; f++) s += xp[f] * wp[f];
    int b = bn >> 10, n = bn & 1023, h = o >> 6, d = o & 63;
    out[(((size_t)(b * 4 + h)) * 1024 + n) * 64 + d] = s;
}

// ---------------- LayerNorm stats: one block per batch index ----------------
__global__ void k_lnstats(const float* __restrict__ src, float* __restrict__ stats, int cnt) {
    __shared__ float ss[256], sq[256];
    int b = blockIdx.x, t = threadIdx.x;
    const float* p = src + (size_t)b * cnt;
    float s = 0.f, q = 0.f;
    for (int i = t; i < cnt; i += 256) { float v = p[i]; s += v; q += v * v; }
    ss[t] = s; sq[t] = q; __syncthreads();
    for (int o = 128; o > 0; o >>= 1) {
        if (t < o) { ss[t] += ss[t + o]; sq[t] += sq[t + o]; }
        __syncthreads();
    }
    if (t == 0) {
        float m = ss[0] / cnt;
        float v = sq[0] / cnt - m * m;
        stats[b * 2] = m; stats[b * 2 + 1] = rsqrtf(v + 1e-5f);
    }
}

__global__ void k_lnapply(float* __restrict__ data, const float* __restrict__ st,
                          const float* __restrict__ gam, const float* __restrict__ bet) {
    int g = blockIdx.x * 256 + threadIdx.x;          // 16*262144
    int b = g >> 18, j = g & (CNT1_ - 1);
    float v = (data[g] - st[b * 2]) * st[b * 2 + 1];
    data[g] = v * gam[j] + bet[j];
}

// ---- pack [Q|K] rows into WMMA A-fragments (16x32 bf16, ISA lane swizzle) ---
__global__ void k_pack_qkA(const float* __restrict__ Qn, const float* __restrict__ Kn,
                           unsigned short* __restrict__ aqk) {
    int g = blockIdx.x * 256 + threadIdx.x;          // BH*64*4*32 = 524288
    int lane = g & 31, kt = (g >> 5) & 3, mt = (g >> 7) & 63, bh = g >> 13;
    int n = mt * 16 + (lane & 15);
    const float* qp = Qn + ((size_t)bh * 1024 + n) * 64;
    const float* kp = Kn + ((size_t)bh * 1024 + n) * 64;
    v16us out;
    #pragma unroll
    for (int e = 0; e < 16; e++) {
        int k = kt * 32 + e + 8 * ((e >> 3) + (lane >> 4));
        out[e] = f2bf(k < 64 ? qp[k] : kp[k - 64]);
    }
    *(v16us*)(aqk + (size_t)g * 16) = out;
}

// ---- pack B-fragments (32x16 bf16): lanes0-15 K=0..7,16..23, lanes16-31 K=8..15,24..31 --
__global__ void k_pack_wcat(const float* __restrict__ qlw, const float* __restrict__ klw,
                            unsigned short* __restrict__ wc) {
    int g = blockIdx.x * 256 + threadIdx.x;          // 4*64*32 = 8192
    int lane = g & 31, nt = (g >> 5) & 63, kt = g >> 11;
    int col = nt * 16 + (lane & 15);
    v16us out;
    #pragma unroll
    for (int e = 0; e < 16; e++) {
        int k = kt * 32 + e + 8 * ((e >> 3) + (lane >> 4));
        out[e] = f2bf(k < 64 ? qlw[(size_t)col * 64 + k] : klw[(size_t)col * 64 + k - 64]);
    }
    *(v16us*)(wc + (size_t)g * 16) = out;
}

__global__ void k_pack_alw(const float* __restrict__ alw, unsigned short* __restrict__ af) {
    int g = blockIdx.x * 256 + threadIdx.x;          // 32*64*32 = 65536
    int lane = g & 31, nt = (g >> 5) & 63, kt = g >> 11;
    int p = nt * 16 + (lane & 15);
    v16us out;
    #pragma unroll
    for (int e = 0; e < 16; e++) {
        int m = kt * 32 + e + 8 * ((e >> 3) + (lane >> 4));
        out[e] = f2bf(alw[(size_t)p * 1024 + m]);
    }
    *(v16us*)(af + (size_t)g * 16) = out;
}

__global__ void k_pack_v(const float* __restrict__ Vn, unsigned short* __restrict__ vf) {
    int g = blockIdx.x * 256 + threadIdx.x;          // BH*32*4*32 = 262144
    int lane = g & 31, nt = (g >> 5) & 3, kt = (g >> 7) & 31, bh = g >> 12;
    int d = nt * 16 + (lane & 15);
    v16us out;
    #pragma unroll
    for (int e = 0; e < 16; e++) {
        int m = kt * 32 + e + 8 * ((e >> 3) + (lane >> 4));
        out[e] = f2bf(Vn[((size_t)bh * 1024 + m) * 64 + d]);
    }
    *(v16us*)(vf + (size_t)g * 16) = out;
}

// ------- GEMM1: S = elu([Q|K]@[qlw|klw]^T + qlb + klb), bf16 row-major ------
// Register-blocked: each wave computes 32x64 (2 M-tiles x 4 N-tiles).
__global__ void k_gemm1(const unsigned short* __restrict__ aqk,
                        const unsigned short* __restrict__ wcat,
                        const float* __restrict__ qlb, const float* __restrict__ klb,
                        unsigned short* __restrict__ a1rm) {
    int wid = blockIdx.x * 8 + (threadIdx.x >> 5);   // 64*32*16 = 32768 waves
    int lane = threadIdx.x & 31;
    int ntb = wid & 15, mtb = (wid >> 4) & 31, bh = wid >> 9;
    v8f acc[2][4] = {};
    const unsigned short* ab = aqk + ((size_t)(bh * 64 + mtb * 2) * 4) * FRAG_ + lane * 16;
    const unsigned short* bb = wcat + (size_t)(ntb * 4) * FRAG_ + lane * 16;
    #pragma unroll
    for (int kt = 0; kt < 4; kt++) {
        v16us a0 = *(const v16us*)(ab + (size_t)kt * FRAG_);
        v16us a1 = *(const v16us*)(ab + (size_t)(4 + kt) * FRAG_);
        v16us bf[4];
        #pragma unroll
        for (int ni = 0; ni < 4; ni++)
            bf[ni] = *(const v16us*)(bb + (size_t)(kt * 64 + ni) * FRAG_);
        #pragma unroll
        for (int ni = 0; ni < 4; ni++) {
            acc[0][ni] = wmma_acc(a0, bf[ni], acc[0][ni]);
            acc[1][ni] = wmma_acc(a1, bf[ni], acc[1][ni]);
        }
    }
    #pragma unroll
    for (int mi = 0; mi < 2; mi++) {
        int mt = mtb * 2 + mi;
        #pragma unroll
        for (int ni = 0; ni < 4; ni++) {
            int m = (ntb * 4 + ni) * 16 + (lane & 15);
            float bias = qlb[m] + klb[m];
            size_t base = ((size_t)bh << 20) + (size_t)(mt * 16 + 8 * (lane >> 4)) * 1024 + m;
            #pragma unroll
            for (int r = 0; r < 8; r++) {
                float v = acc[mi][ni][r] + bias;
                v = v > 0.f ? v : (__expf(v) - 1.0f);       // elu
                a1rm[base + (size_t)r * 1024] = f2bf(v);
            }
        }
    }
}

// ------ repack S row-major -> WMMA A-fragments for GEMM2 --------------------
__global__ void k_repackA1(const unsigned short* __restrict__ a1rm,
                           unsigned short* __restrict__ a1fr) {
    int g = blockIdx.x * 256 + threadIdx.x;          // BH*64*32*32 = 4194304
    int lane = g & 31, kt = (g >> 5) & 31, mt = (g >> 10) & 63, bh = g >> 16;
    int n = mt * 16 + (lane & 15);
    const unsigned short* rp = a1rm + ((size_t)bh << 20) + (size_t)n * 1024 + kt * 32;
    v16us out;
    #pragma unroll
    for (int e = 0; e < 16; e++) out[e] = rp[e + 8 * ((e >> 3) + (lane >> 4))];
    *(v16us*)(a1fr + (size_t)g * 16) = out;
}

// ------ GEMM2 (dominant, 137 GFLOP): A2 = S@alw^T + alb, bf16 row-major -----
// Register-blocked 32x64 per wave: 12 b128 loads feed 8 WMMAs per k-step.
__global__ void k_gemm2(const unsigned short* __restrict__ a1fr,
                        const unsigned short* __restrict__ alwf,
                        const float* __restrict__ alb,
                        unsigned short* __restrict__ a2rm) {
    int wid = blockIdx.x * 8 + (threadIdx.x >> 5);   // 64*32*16 = 32768 waves
    int lane = threadIdx.x & 31;
    int ntb = wid & 15, mtb = (wid >> 4) & 31, bh = wid >> 9;
    v8f acc[2][4] = {};
    const unsigned short* ab = a1fr + ((size_t)(bh * 64 + mtb * 2) * 32) * FRAG_ + lane * 16;
    const unsigned short* bb = alwf + (size_t)(ntb * 4) * FRAG_ + lane * 16;
    for (int kt = 0; kt < 32; kt++) {
        v16us a0 = *(const v16us*)(ab + (size_t)kt * FRAG_);
        v16us a1 = *(const v16us*)(ab + (size_t)(32 + kt) * FRAG_);
        v16us bf[4];
        #pragma unroll
        for (int ni = 0; ni < 4; ni++)
            bf[ni] = *(const v16us*)(bb + (size_t)(kt * 64 + ni) * FRAG_);
        #pragma unroll
        for (int ni = 0; ni < 4; ni++) {
            acc[0][ni] = wmma_acc(a0, bf[ni], acc[0][ni]);
            acc[1][ni] = wmma_acc(a1, bf[ni], acc[1][ni]);
        }
    }
    #pragma unroll
    for (int mi = 0; mi < 2; mi++) {
        int mt = mtb * 2 + mi;
        #pragma unroll
        for (int ni = 0; ni < 4; ni++) {
            int p = (ntb * 4 + ni) * 16 + (lane & 15);
            float bias = alb[p];
            size_t base = ((size_t)bh << 20) + (size_t)(mt * 16 + 8 * (lane >> 4)) * 1024 + p;
            #pragma unroll
            for (int r = 0; r < 8; r++)
                a2rm[base + (size_t)r * 1024] = f2bf(acc[mi][ni][r] + bias);
        }
    }
}

// ------ softmax over p (one block/row) and scatter into A-fragments ---------
__global__ void k_softmax_pack(const unsigned short* __restrict__ a2,
                               unsigned short* __restrict__ pf) {
    __shared__ float red[256];
    int row = blockIdx.x;                            // bh*1024 + n
    int t = threadIdx.x;
    int bh = row >> 10, n = row & 1023;
    const unsigned short* rp = a2 + (size_t)row * 1024;
    float x[4], mx = -3.0e38f;
    #pragma unroll
    for (int i = 0; i < 4; i++) { x[i] = bf2f(rp[t + i * 256]); mx = fmaxf(mx, x[i]); }
    red[t] = mx; __syncthreads();
    for (int o = 128; o > 0; o >>= 1) { if (t < o) red[t] = fmaxf(red[t], red[t + o]); __syncthreads(); }
    mx = red[0]; __syncthreads();
    float s = 0.f;
    #pragma unroll
    for (int i = 0; i < 4; i++) { x[i] = __expf(x[i] - mx); s += x[i]; }
    red[t] = s; __syncthreads();
    for (int o = 128; o > 0; o >>= 1) { if (t < o) red[t] += red[t + o]; __syncthreads(); }
    float inv = 1.0f / red[0];
    int mt = n >> 4;
    #pragma unroll
    for (int i = 0; i < 4; i++) {
        int m = t + i * 256;
        int kt = m >> 5, kk = m & 31, q = kk >> 3;
        int h = q & 1, e = (kk & 7) + ((q >> 1) << 3);
        int lane = (h << 4) | (n & 15);
        size_t frag = (size_t)(bh * 64 + mt) * 32 + kt;
        pf[frag * FRAG_ + lane * 16 + e] = f2bf(x[i] * inv);
    }
}

// ------ GEMM3: E = P@V, f32 output directly into [B,N,HEADS*D] layout -------
// Register-blocked: each wave computes 32x64 (entire D width of one head).
__global__ void k_gemm3(const unsigned short* __restrict__ pf,
                        const unsigned short* __restrict__ vf,
                        float* __restrict__ ecat) {
    int wid = blockIdx.x * 8 + (threadIdx.x >> 5);   // 64*32 = 2048 waves
    int lane = threadIdx.x & 31;
    int mtb = wid & 31, bh = wid >> 5;
    v8f acc[2][4] = {};
    const unsigned short* ab = pf + ((size_t)(bh * 64 + mtb * 2) * 32) * FRAG_ + lane * 16;
    const unsigned short* bb = vf + ((size_t)(bh * 32) * 4) * FRAG_ + lane * 16;
    for (int kt = 0; kt < 32; kt++) {
        v16us a0 = *(const v16us*)(ab + (size_t)kt * FRAG_);
        v16us a1 = *(const v16us*)(ab + (size_t)(32 + kt) * FRAG_);
        v16us bf[4];
        #pragma unroll
        for (int ni = 0; ni < 4; ni++)
            bf[ni] = *(const v16us*)(bb + (size_t)(kt * 4 + ni) * FRAG_);
        #pragma unroll
        for (int ni = 0; ni < 4; ni++) {
            acc[0][ni] = wmma_acc(a0, bf[ni], acc[0][ni]);
            acc[1][ni] = wmma_acc(a1, bf[ni], acc[1][ni]);
        }
    }
    int b = bh >> 2, h = bh & 3;
    #pragma unroll
    for (int mi = 0; mi < 2; mi++) {
        int mt = mtb * 2 + mi;
        #pragma unroll
        for (int ni = 0; ni < 4; ni++) {
            int c = h * 64 + ni * 16 + (lane & 15);
            size_t base = ((size_t)b * 1024 + mt * 16 + 8 * (lane >> 4)) * 256 + c;
            #pragma unroll
            for (int r = 0; r < 8; r++) ecat[base + (size_t)r * 256] = acc[mi][ni][r];
        }
    }
}

// ------ output FC + ReLU ----------------------------------------------------
__global__ void k_fc(const float* __restrict__ ecat, const float* __restrict__ w,
                     const float* __restrict__ b, float* __restrict__ o64) {
    int g = blockIdx.x * 256 + threadIdx.x;          // 16*1024*64
    int j = g & 63, bn = g >> 6;
    const float* ep = ecat + (size_t)bn * 256;
    const float* wp = w + (size_t)j * 256;
    float s = b[j];
    #pragma unroll 8
    for (int c = 0; c < 256; c++) s += ep[c] * wp[c];
    o64[g] = fmaxf(s, 0.f);
}

// ------ final LN (no affine) + max over N -----------------------------------
__global__ void k_lnmax(const float* __restrict__ o64, const float* __restrict__ st,
                        float* __restrict__ out) {
    int b = blockIdx.x, j = threadIdx.x;             // 16 x 64
    float m = st[b * 2], r = st[b * 2 + 1];
    const float* p = o64 + (size_t)b * 65536 + j;
    float best = -3.0e38f;
    for (int n = 0; n < 1024; n++) best = fmaxf(best, (p[(size_t)n * 64] - m) * r);
    out[b * 64 + j] = best;
}

extern "C" void kernel_launch(void* const* d_in, const int* in_sizes, int n_in,
                              void* d_out, int out_size, void* d_ws, size_t ws_size,
                              hipStream_t stream) {
    (void)in_sizes; (void)n_in; (void)out_size; (void)ws_size;
    const float* x   = (const float*)d_in[0];
    const float* kpw = (const float*)d_in[1];  const float* kpb = (const float*)d_in[2];
    const float* qpw = (const float*)d_in[3];  const float* qpb = (const float*)d_in[4];
    const float* vpw = (const float*)d_in[5];  const float* vpb = (const float*)d_in[6];
    const float* kng = (const float*)d_in[7];  const float* knb = (const float*)d_in[8];
    const float* qng = (const float*)d_in[9];  const float* qnb = (const float*)d_in[10];
    const float* vng = (const float*)d_in[11]; const float* vnb = (const float*)d_in[12];
    const float* klw = (const float*)d_in[13]; const float* klb = (const float*)d_in[14];
    const float* qlw = (const float*)d_in[15]; const float* qlb = (const float*)d_in[16];
    const float* alw = (const float*)d_in[17]; const float* alb = (const float*)d_in[18];
    const float* l1w = (const float*)d_in[19]; const float* l1b = (const float*)d_in[20];
    float* out = (float*)d_out;

    // ---- workspace layout (≈369 MB, with aliased reuse) ----
    char* w = (char*)d_ws;
    float* XIN = (float*)w;              w += (size_t)16384 * FEAT_ * 4;        // 1.7 MB
    float* KR  = (float*)w;              w += (size_t)B_ * CNT1_ * 4;           // 16.8 MB
    float* QR  = (float*)w;              w += (size_t)B_ * CNT1_ * 4;
    float* VR  = (float*)w;              w += (size_t)B_ * CNT1_ * 4;
    float* ST  = (float*)w;              w += 512;                              // 3 x 16 x {mean,rstd}
    unsigned short* AQK = (unsigned short*)w; w += (size_t)BH_ * 64 * 4 * FRAG_ * 2;   // 16.8 MB
    unsigned short* WCF = (unsigned short*)w; w += (size_t)4 * 64 * FRAG_ * 2;         // 0.26 MB
    unsigned short* ALF = (unsigned short*)w; w += (size_t)32 * 64 * FRAG_ * 2;        // 2 MB
    unsigned short* VF  = (unsigned short*)w; w += (size_t)BH_ * 32 * 4 * FRAG_ * 2;   // 8.4 MB
    unsigned short* A1RM = (unsigned short*)w; w += (size_t)BH_ * 1024 * 1024 * 2;     // 134 MB (reused as A2RM)
    unsigned short* A1F  = (unsigned short*)w; w += (size_t)BH_ * 1024 * 1024 * 2;     // 134 MB (reused as PFRAG)
    float* ECAT = (float*)w;             w += (size_t)B_ * 1024 * 256 * 4;      // 16.8 MB
    float* O64  = (float*)w;             w += (size_t)B_ * 1024 * 64 * 4;       // 4.2 MB
    float* ST2  = (float*)w;             w += 512;

    // 0) input assembly + projections
    k_xin<<<64, 256, 0, stream>>>(x, XIN);
    k_proj<<<16384, 256, 0, stream>>>(XIN, kpw, kpb, KR);
    k_proj<<<16384, 256, 0, stream>>>(XIN, qpw, qpb, QR);
    k_proj<<<16384, 256, 0, stream>>>(XIN, vpw, vpb, VR);
    // 1) per-batch LayerNorms (in place)
    k_lnstats<<<16, 256, 0, stream>>>(KR, ST +  0, CNT1_);
    k_lnstats<<<16, 256, 0, stream>>>(QR, ST + 32, CNT1_);
    k_lnstats<<<16, 256, 0, stream>>>(VR, ST + 64, CNT1_);
    k_lnapply<<<16384, 256, 0, stream>>>(KR, ST +  0, kng, knb);
    k_lnapply<<<16384, 256, 0, stream>>>(QR, ST + 32, qng, qnb);
    k_lnapply<<<16384, 256, 0, stream>>>(VR, ST + 64, vng, vnb);
    // 2) fragment packing (bf16, WMMA-native layouts)
    k_pack_qkA<<<2048, 256, 0, stream>>>(QR, KR, AQK);
    k_pack_wcat<<<32, 256, 0, stream>>>(qlw, klw, WCF);
    k_pack_alw<<<256, 256, 0, stream>>>(alw, ALF);
    k_pack_v<<<1024, 256, 0, stream>>>(VR, VF);
    // 3) attention GEMM chain (register-blocked v_wmma_f32_16x16x32_bf16)
    k_gemm1<<<4096, 256, 0, stream>>>(AQK, WCF, qlb, klb, A1RM);
    k_repackA1<<<16384, 256, 0, stream>>>(A1RM, A1F);
    k_gemm2<<<4096, 256, 0, stream>>>(A1F, ALF, alb, A1RM /* = A2RM */);
    k_softmax_pack<<<65536, 256, 0, stream>>>(A1RM, A1F /* = PFRAG */);
    k_gemm3<<<256, 256, 0, stream>>>(A1F, VF, ECAT);
    // 4) output head
    k_fc<<<4096, 256, 0, stream>>>(ECAT, l1w, l1b, O64);
    k_lnstats<<<16, 256, 0, stream>>>(O64, ST2, CNT2_);
    k_lnmax<<<16, 64, 0, stream>>>(O64, ST2, out);
}
